// E2E_model_13134009991200
// MI455X (gfx1250) — matrix operand order repaired and checked by reading, not compiled, not run on hardware
//
#include <hip/hip_runtime.h>
#include <hip/hip_bf16.h>

// ---- model dims ----
#define HIDN   512
#define MIDN   256
#define CONDD  256
#define OPD    32
#define EXD    32
#define LLV    16
#define NNV    1024
#define TTV    8
#define BBV    (LLV * NNV)        // 16384
#define G4H    (4 * HIDN)         // 2048
#define K1     (CONDD + HIDN)     // 768
#define IN2    (OPD + EXD + MIDN) // 320
#define EPSV   1e-5f

typedef __bf16 bf16;
typedef __attribute__((ext_vector_type(16))) __bf16 v16bf;
typedef __attribute__((ext_vector_type(8)))  __bf16 v8bf;
typedef __attribute__((ext_vector_type(8)))  float  v8f;

// =====================================================================
// WMMA GEMM:  C[m,n] = sum_k A[m,k]*W[n,k] + bias[n] (+ add[m,n] if HAS_ADD)
// A: bf16 [M x K] row stride lda ; W: bf16 [N x K] row stride ldw
// Tile per wave: (16*MT) rows x 64 cols.
// grid.x = Nout/64, grid.y = M/(16*MT), blockDim = 32
// =====================================================================
template <int MT, bool HAS_ADD>
__global__ __launch_bounds__(32)
void wmma_gemm_bf16(const bf16* __restrict__ A, int lda,
                    const bf16* __restrict__ W, int ldw,
                    const float* __restrict__ bias,
                    const float* __restrict__ add, long long addStride,
                    float* __restrict__ C, int ldc, int K)
{
    const int lane = threadIdx.x & 31;
    const int r16  = lane & 15;
    const int hi   = lane >> 4;              // lane group 0/1
    const int nBase = blockIdx.x * 64;
    const int mBase = blockIdx.y * (16 * MT);

    v8f acc[MT][4];
    #pragma unroll
    for (int i = 0; i < MT; ++i)
        #pragma unroll
        for (int j = 0; j < 4; ++j)
            acc[i][j] = (v8f){0.f,0.f,0.f,0.f,0.f,0.f,0.f,0.f};

    const bf16* aRow[MT];
    #pragma unroll
    for (int i = 0; i < MT; ++i)
        aRow[i] = A + (size_t)(mBase + 16 * i + r16) * lda;
    const bf16* wRow0 = W + (size_t)(nBase +  0 + r16) * ldw;
    const bf16* wRow1 = W + (size_t)(nBase + 16 + r16) * ldw;
    const bf16* wRow2 = W + (size_t)(nBase + 32 + r16) * ldw;
    const bf16* wRow3 = W + (size_t)(nBase + 48 + r16) * ldw;

    for (int k0 = 0; k0 < K; k0 += 32) {
        // A fragment: lanes 0-15 -> K = k0..k0+15, lanes 16-31 -> K = k0+16..k0+31 (matches B)
        v16bf af[MT];
        #pragma unroll
        for (int i = 0; i < MT; ++i) {
            af[i] = *(const v16bf*)(aRow[i] + k0 + hi * 16);
        }

        // B fragments (K x 16): lane half selects 16 contiguous K values
        v16bf b0 = *(const v16bf*)(wRow0 + k0 + hi * 16);
        v16bf b1 = *(const v16bf*)(wRow1 + k0 + hi * 16);
        v16bf b2 = *(const v16bf*)(wRow2 + k0 + hi * 16);
        v16bf b3 = *(const v16bf*)(wRow3 + k0 + hi * 16);

        #pragma unroll
        for (int i = 0; i < MT; ++i) {
            acc[i][0] = __builtin_amdgcn_wmma_f32_16x16x32_bf16(false, af[i], false, b0, (short)0, acc[i][0], false, false);
            acc[i][1] = __builtin_amdgcn_wmma_f32_16x16x32_bf16(false, af[i], false, b1, (short)0, acc[i][1], false, false);
            acc[i][2] = __builtin_amdgcn_wmma_f32_16x16x32_bf16(false, af[i], false, b2, (short)0, acc[i][2], false, false);
            acc[i][3] = __builtin_amdgcn_wmma_f32_16x16x32_bf16(false, af[i], false, b3, (short)0, acc[i][3], false, false);
        }
    }

    // bias for this lane's 4 column positions (uniform across rows)
    float bcol[4];
    #pragma unroll
    for (int j = 0; j < 4; ++j) bcol[j] = bias[nBase + 16 * j + r16];

    // D layout: VGPR r, lanes 0-15 -> row r, lanes 16-31 -> row r+8; col = lane&15
    #pragma unroll
    for (int i = 0; i < MT; ++i) {
        #pragma unroll
        for (int r = 0; r < 8; ++r) {
            const int m = mBase + 16 * i + r + 8 * hi;
            float* crow = C + (size_t)m * ldc + nBase + r16;
            const float* arow = HAS_ADD ? (add + (size_t)m * addStride + nBase + r16) : nullptr;
            #pragma unroll
            for (int j = 0; j < 4; ++j) {
                float v = acc[i][j][r] + bcol[j];
                if (HAS_ADD) v += arow[16 * j];
                crow[16 * j] = v;
            }
        }
    }
}

// =====================================================================
// elementwise / prep kernels
// =====================================================================
__global__ __launch_bounds__(256)
void zero_bf16_kernel(bf16* p, int n) {
    int t = blockIdx.x * blockDim.x + threadIdx.x;
    if (t < n) p[t] = (bf16)0.f;
}
__global__ __launch_bounds__(256)
void zero_f32_kernel(float* p, int n) {
    int t = blockIdx.x * blockDim.x + threadIdx.x;
    if (t < n) p[t] = 0.f;
}
__global__ __launch_bounds__(256)
void cvt_f32_bf16_kernel(const float* __restrict__ in, bf16* __restrict__ out, int n) {
    int t = blockIdx.x * blockDim.x + threadIdx.x;
    if (t < n) out[t] = (bf16)in[t];
}
// Wcat1[g, 0:256] = lstm1_Wih[g,:], Wcat1[g, 256:768] = lstm1_Whh[g,:]
__global__ __launch_bounds__(256)
void wcat1_kernel(const float* __restrict__ Wih, const float* __restrict__ Whh,
                  bf16* __restrict__ out) {
    int t = blockIdx.x * blockDim.x + threadIdx.x;
    if (t >= G4H * K1) return;
    int g = t / K1, c = t - g * K1;
    float v = (c < CONDD) ? Wih[(size_t)g * CONDD + c] : Whh[(size_t)g * HIDN + (c - CONDD)];
    out[t] = (bf16)v;
}
__global__ __launch_bounds__(256)
void bsum_kernel(const float* a, const float* b, float* out, int n) {
    int t = blockIdx.x * blockDim.x + threadIdx.x;
    if (t < n) out[t] = a[t] + b[t];
}
// Xcat2[:, 0:32]=operators, [:, 32:64]=extra_infos (bf16)
__global__ __launch_bounds__(256)
void opex_kernel(const float* __restrict__ ops, const float* __restrict__ ex,
                 bf16* __restrict__ xcat) {
    int t = blockIdx.x * blockDim.x + threadIdx.x;
    if (t >= BBV * (OPD + EXD)) return;
    int r = t / (OPD + EXD), c = t - r * (OPD + EXD);
    float v = (c < OPD) ? ops[(size_t)r * OPD + c] : ex[(size_t)r * EXD + (c - OPD)];
    xcat[(size_t)r * IN2 + c] = (bf16)v;
}
// Acat[:, 0:256] = bf16(conditions[:, tstep, :])
__global__ __launch_bounds__(256)
void xcopy_kernel(const float* __restrict__ cond, bf16* __restrict__ acat, int tstep) {
    int t = blockIdx.x * blockDim.x + threadIdx.x;
    if (t >= BBV * CONDD) return;
    int b = t >> 8, d = t & (CONDD - 1);
    acat[(size_t)b * K1 + d] = (bf16)cond[((size_t)b * TTV + tstep) * CONDD + d];
}
// LSTM cell: gates fp32 [M,4H] (i,f,g,o), c update, h out (bf16 strided / f32)
__global__ __launch_bounds__(256)
void lstm_cell_kernel(const float* __restrict__ G,
                      const float* __restrict__ c_in, float* __restrict__ c_out,
                      bf16* __restrict__ h_bf16, int h_stride,
                      float* __restrict__ h_f32, int M, int H)
{
    int t = blockIdx.x * blockDim.x + threadIdx.x;
    if (t >= M * H) return;
    int m = t / H, d = t - m * H;
    size_t gb = (size_t)m * 4 * H;
    float gi = G[gb + d];
    float gf = G[gb + H + d];
    float gg = G[gb + 2 * H + d];
    float go = G[gb + 3 * H + d];
    gi = 1.f / (1.f + __expf(-gi));
    gf = 1.f / (1.f + __expf(-gf));
    gg = tanhf(gg);
    go = 1.f / (1.f + __expf(-go));
    float c = gf * c_in[(size_t)m * H + d] + gi * gg;
    float h = go * tanhf(c);
    c_out[(size_t)m * H + d] = c;
    if (h_bf16) h_bf16[(size_t)m * h_stride + d] = (bf16)h;
    if (h_f32)  h_f32[(size_t)m * H + d] = h;
}
// tree gather: hg = bf16(0.5*(ph[m0]+ph[m1])), cg = 0.5*(pc[m0]+pc[m1])
__global__ __launch_bounds__(256)
void tree_gather_kernel(const long long* __restrict__ mapping,
                        const float* __restrict__ h, const float* __restrict__ c,
                        bf16* __restrict__ hg, float* __restrict__ cg)
{
    int t = blockIdx.x * blockDim.x + threadIdx.x;
    if (t >= NNV * HIDN) return;
    int n = t >> 9, d = t & (HIDN - 1);
    long long m0 = mapping[2 * n], m1 = mapping[2 * n + 1];
    float hv = 0.f, cv = 0.f;
    if (m0 > 0) { hv += h[(size_t)(m0 - 1) * HIDN + d]; cv += c[(size_t)(m0 - 1) * HIDN + d]; }
    if (m1 > 0) { hv += h[(size_t)(m1 - 1) * HIDN + d]; cv += c[(size_t)(m1 - 1) * HIDN + d]; }
    hg[t] = (bf16)(0.5f * hv);
    cg[t] = 0.5f * cv;
}
// per-column sum / sumsq (optionally after relu); one block per column
__global__ __launch_bounds__(256)
void col_stats_kernel(const float* __restrict__ X, int rows, int cols,
                      int doRelu, float* __restrict__ stats)
{
    __shared__ float s[512];
    int col = blockIdx.x;
    float a = 0.f, b = 0.f;
    for (int r = threadIdx.x; r < rows; r += blockDim.x) {
        float v = X[(size_t)r * cols + col];
        if (doRelu) v = fmaxf(v, 0.f);
        a += v; b += v * v;
    }
    s[threadIdx.x] = a; s[256 + threadIdx.x] = b;
    __syncthreads();
    for (int off = 128; off > 0; off >>= 1) {
        if (threadIdx.x < (unsigned)off) {
            s[threadIdx.x] += s[threadIdx.x + off];
            s[256 + threadIdx.x] += s[256 + threadIdx.x + off];
        }
        __syncthreads();
    }
    if (threadIdx.x == 0) { stats[col] = s[0]; stats[cols + col] = s[256]; }
}
// BN1: relu + normalize, write bf16 into Xcat2 cols [64:320)
__global__ __launch_bounds__(256)
void bn1_norm_kernel(const float* __restrict__ X, const float* __restrict__ stats,
                     const float* __restrict__ gamma, const float* __restrict__ beta,
                     bf16* __restrict__ xcat)
{
    int t = blockIdx.x * blockDim.x + threadIdx.x;
    if (t >= BBV * MIDN) return;
    int r = t >> 8, j = t & (MIDN - 1);
    float v = fmaxf(X[(size_t)r * MIDN + j], 0.f);
    const float inv = 1.f / (float)BBV;
    float mu = stats[j] * inv;
    float var = stats[MIDN + j] * inv - mu * mu;
    float y = gamma[j] * (v - mu) * rsqrtf(var + EPSV) + beta[j];
    xcat[(size_t)r * IN2 + (OPD + EXD) + j] = (bf16)y;
}
// BN2: normalize h2 [N,HID] -> d_out fp32
__global__ __launch_bounds__(256)
void bn2_norm_kernel(const float* __restrict__ X, const float* __restrict__ stats,
                     const float* __restrict__ gamma, const float* __restrict__ beta,
                     float* __restrict__ out)
{
    int t = blockIdx.x * blockDim.x + threadIdx.x;
    if (t >= NNV * HIDN) return;
    int j = t & (HIDN - 1);
    const float inv = 1.f / (float)NNV;
    float mu = stats[j] * inv;
    float var = stats[HIDN + j] * inv - mu * mu;
    out[t] = gamma[j] * (X[t] - mu) * rsqrtf(var + EPSV) + beta[j];
}

// =====================================================================
// workspace layout (bytes, all 256-aligned)
// =====================================================================
#define WS_ACAT    ((size_t)0)                       // bf16 [B,768]      25165824
#define WS_G       ((size_t)25165824)                // f32  [B,2048]     134217728 (reused as xp2)
#define WS_C1      ((size_t)159383552)               // f32  [B,512]      33554432
#define WS_WCAT1   ((size_t)192937984)               // bf16 [2048,768]   3145728
#define WS_BSUM1   ((size_t)196083712)               // f32  [2048]       8192
#define WS_CONDWB  ((size_t)196091904)               // bf16 [256,512]    262144
#define WS_LASTPRE ((size_t)196354048)               // f32  [B,256]      16777216
#define WS_STATS1  ((size_t)213131264)               // f32  [512]        2048
#define WS_XCAT2   ((size_t)213133312)               // bf16 [B,320]      10485760
#define WS_W2IHB   ((size_t)223619072)               // bf16 [2048,320]   1310720
#define WS_W2HHB   ((size_t)224929792)               // bf16 [2048,512]   2097152
#define WS_HG      ((size_t)227026944)               // bf16 [N,512]      1048576
#define WS_CG      ((size_t)228075520)               // f32  [N,512]      2097152
#define WS_H2      ((size_t)230172672)               // f32  [N,512]      2097152
#define WS_C2      ((size_t)232269824)               // f32  [N,512]      2097152
#define WS_G2      ((size_t)234366976)               // f32  [N,2048]     8388608
#define WS_STATS2  ((size_t)242755584)               // f32  [1024]       4096

static inline int cdiv(int a, int b) { return (a + b - 1) / b; }
#define EW(kern, n, ...) kern<<<dim3(cdiv((n), 256)), dim3(256), 0, stream>>>(__VA_ARGS__)

extern "C" void kernel_launch(void* const* d_in, const int* in_sizes, int n_in,
                              void* d_out, int out_size, void* d_ws, size_t ws_size,
                              hipStream_t stream)
{
    const float*     operators   = (const float*)d_in[0];
    const float*     extra_infos = (const float*)d_in[1];
    const float*     conditions  = (const float*)d_in[2];
    // d_in[3] condition_masks: unused by reference
    const long long* mapping     = (const long long*)d_in[4];
    const float* lstm1_Wih = (const float*)d_in[5];
    const float* lstm1_Whh = (const float*)d_in[6];
    const float* lstm1_bih = (const float*)d_in[7];
    const float* lstm1_bhh = (const float*)d_in[8];
    const float* cond_W    = (const float*)d_in[9];
    const float* cond_b    = (const float*)d_in[10];
    const float* bn1_g     = (const float*)d_in[11];
    const float* bn1_b     = (const float*)d_in[12];
    const float* lstm2_Wih = (const float*)d_in[13];
    const float* lstm2_Whh = (const float*)d_in[14];
    const float* lstm2_bih = (const float*)d_in[15];
    const float* lstm2_bhh = (const float*)d_in[16];
    const float* bn2_g     = (const float*)d_in[17];
    const float* bn2_b     = (const float*)d_in[18];

    char* ws = (char*)d_ws;
    bf16*  Acat    = (bf16*)(ws + WS_ACAT);
    float* G       = (float*)(ws + WS_G);
    float* xp2     = G;                       // reuse after LSTM1
    float* C1      = (float*)(ws + WS_C1);
    bf16*  Wcat1   = (bf16*)(ws + WS_WCAT1);
    float* bsum1   = (float*)(ws + WS_BSUM1);
    bf16*  condWb  = (bf16*)(ws + WS_CONDWB);
    float* lastpre = (float*)(ws + WS_LASTPRE);
    float* stats1  = (float*)(ws + WS_STATS1);
    bf16*  Xcat2   = (bf16*)(ws + WS_XCAT2);
    bf16*  W2ihb   = (bf16*)(ws + WS_W2IHB);
    bf16*  W2hhb   = (bf16*)(ws + WS_W2HHB);
    bf16*  hg      = (bf16*)(ws + WS_HG);
    float* cg      = (float*)(ws + WS_CG);
    float* h2      = (float*)(ws + WS_H2);
    float* c2      = (float*)(ws + WS_C2);
    float* G2      = (float*)(ws + WS_G2);
    float* stats2  = (float*)(ws + WS_STATS2);
    float* out     = (float*)d_out;

    // ---- weight prep (bf16) ----
    EW(wcat1_kernel, G4H * K1, lstm1_Wih, lstm1_Whh, Wcat1);
    EW(bsum_kernel, G4H, lstm1_bih, lstm1_bhh, bsum1, G4H);
    EW(cvt_f32_bf16_kernel, MIDN * HIDN, cond_W, condWb, MIDN * HIDN);
    EW(cvt_f32_bf16_kernel, G4H * IN2, lstm2_Wih, W2ihb, G4H * IN2);
    EW(cvt_f32_bf16_kernel, G4H * HIDN, lstm2_Whh, W2hhb, G4H * HIDN);
    EW(opex_kernel, BBV * (OPD + EXD), operators, extra_infos, Xcat2);

    // ---- LSTM1 over T=8 steps, fused [x_t | h] @ [Wih | Whh]^T ----
    EW(zero_bf16_kernel, BBV * K1, Acat, BBV * K1);
    EW(zero_f32_kernel, BBV * HIDN, C1, BBV * HIDN);
    for (int t = 0; t < TTV; ++t) {
        EW(xcopy_kernel, BBV * CONDD, conditions, Acat, t);
        wmma_gemm_bf16<2, false><<<dim3(G4H / 64, BBV / 32), dim3(32), 0, stream>>>(
            Acat, K1, Wcat1, K1, bsum1, nullptr, 0, G, G4H, K1);
        EW(lstm_cell_kernel, BBV * HIDN, G, C1, C1, Acat + CONDD, K1, nullptr, BBV, HIDN);
    }

    // ---- cond linear + relu + BN1 -> Xcat2[:,64:320) ----
    wmma_gemm_bf16<2, false><<<dim3(MIDN / 64, BBV / 32), dim3(32), 0, stream>>>(
        Acat + CONDD, K1, condWb, HIDN, cond_b, nullptr, 0, lastpre, MIDN, HIDN);
    col_stats_kernel<<<dim3(MIDN), dim3(256), 0, stream>>>(lastpre, BBV, MIDN, 1, stats1);
    EW(bn1_norm_kernel, BBV * MIDN, lastpre, stats1, bn1_g, bn1_b, Xcat2);

    // ---- xp2 = Xcat2 @ W2ih^T + bih2  (reuses G buffer) ----
    wmma_gemm_bf16<2, false><<<dim3(G4H / 64, BBV / 32), dim3(32), 0, stream>>>(
        Xcat2, IN2, W2ihb, IN2, lstm2_bih, nullptr, 0, xp2, G4H, IN2);

    // ---- LSTM2 tree: level 15 with zero state, then 14..0 with gathered state ----
    EW(zero_bf16_kernel, NNV * HIDN, hg, NNV * HIDN);
    EW(zero_f32_kernel, NNV * HIDN, cg, NNV * HIDN);
    {
        const float* addp = xp2 + (size_t)(LLV - 1) * NNV * G4H;
        wmma_gemm_bf16<1, true><<<dim3(G4H / 64, NNV / 16), dim3(32), 0, stream>>>(
            hg, HIDN, W2hhb, HIDN, lstm2_bhh, addp, G4H, G2, G4H, HIDN);
        EW(lstm_cell_kernel, NNV * HIDN, G2, cg, c2, (bf16*)nullptr, 0, h2, NNV, HIDN);
    }
    for (int lvl = LLV - 2; lvl >= 0; --lvl) {
        const long long* mp = mapping + (size_t)lvl * NNV * 2;
        EW(tree_gather_kernel, NNV * HIDN, mp, h2, c2, hg, cg);
        const float* addp = xp2 + (size_t)lvl * NNV * G4H;
        wmma_gemm_bf16<1, true><<<dim3(G4H / 64, NNV / 16), dim3(32), 0, stream>>>(
            hg, HIDN, W2hhb, HIDN, lstm2_bhh, addp, G4H, G2, G4H, HIDN);
        EW(lstm_cell_kernel, NNV * HIDN, G2, cg, c2, (bf16*)nullptr, 0, h2, NNV, HIDN);
    }

    // ---- BN2 -> output ----
    col_stats_kernel<<<dim3(HIDN), dim3(256), 0, stream>>>(h2, NNV, HIDN, 0, stats2);
    EW(bn2_norm_kernel, NNV * HIDN, h2, stats2, bn2_g, bn2_b, out);
}
